// ScannedRNN_12721693130902
// MI455X (gfx1250) — matrix operand order repaired and checked by reading, not compiled, not run on hardware
//
#include <hip/hip_runtime.h>
#include <hip/hip_bf16.h>

// GRU scan for MI455X (gfx1250): persistent per-N-tile workgroups, bf16 WMMA
// 16x16x32 with fp32 accumulation, all weights LDS-resident (bf16, ~207KB of
// the 320KB/WGP budget), h carried in fp32 registers across the 1024-step scan.

typedef __attribute__((ext_vector_type(16))) __bf16 v16bf;
typedef __attribute__((ext_vector_type(8)))  __bf16 v8bf;
typedef __attribute__((ext_vector_type(4)))  __bf16 v4bf;
typedef __attribute__((ext_vector_type(8)))  float  v8f;

#define T_STEPS 1024
#define N_ENV   256
#define D_OBS   128
#define F_FEAT  128

__device__ __forceinline__ v8f wmma_bf16(v16bf a, v16bf b, v8f c) {
  // D = A(16x32 bf16) * B(32x16 bf16) + C(16x16 f32)
  return __builtin_amdgcn_wmma_f32_16x16x32_bf16(
      /*neg_a=*/false, a, /*neg_b=*/false, b,
      /*c_mod=*/(short)0, c, /*reuse_a=*/false, /*reuse_b=*/false);
}

// A-fragment (16x32 bf16) from LDS row-major [16][128] buffer, K-chunk c.
// ISA layout: lanes 0-15 row M=lane, K = {c*32+0..7, c*32+16..23};
//             lanes 16-31 row M=lane-16, K = {c*32+8..15, c*32+24..31}.
__device__ __forceinline__ v16bf load_a_frag(const __bf16* buf, int lane, int c) {
  const int m   = lane & 15;
  const int off = (lane >> 4) * 8;
  const __bf16* p = buf + m * 128 + c * 32 + off;
  v8bf lo = *(const v8bf*)(p);
  v8bf hi = *(const v8bf*)(p + 16);
  v16bf f;
#pragma unroll
  for (int i = 0; i < 8; ++i) { f[i] = lo[i]; f[i + 8] = hi[i]; }
  return f;
}

// B-fragment (32x16 bf16) from LDS column-major weight buffer wT[col][k],
// column tile base col0, K-chunk c.
// ISA layout: lanes 0-15 col N=lane hold K = {c*32+0..7, c*32+16..23};
//             lanes 16-31 hold K = {c*32+8..15, c*32+24..31}.
__device__ __forceinline__ v16bf load_b_frag(const __bf16* wT, int col0, int lane, int c) {
  const int n   = lane & 15;
  const int off = (lane >> 4) * 8;
  const __bf16* p = wT + (col0 + n) * 128 + c * 32 + off;
  v8bf b0 = *(const v8bf*)(p);
  v8bf b1 = *(const v8bf*)(p + 16);
  v16bf f;
#pragma unroll
  for (int i = 0; i < 8; ++i) { f[i] = b0[i]; f[i + 8] = b1[i]; }
  return f;
}

__global__ __launch_bounds__(256)
void gru_scan_wmma_kernel(const float* __restrict__ h0,
                          const float* __restrict__ obs,
                          const int*   __restrict__ resets,
                          const float* __restrict__ Wi,
                          const float* __restrict__ bi,
                          const float* __restrict__ Whr,
                          const float* __restrict__ Whz,
                          const float* __restrict__ Whn,
                          const float* __restrict__ bhn,
                          float* __restrict__ out) {
  constexpr int T = T_STEPS, N = N_ENV, D = D_OBS, F = F_FEAT;

  // LDS: weights (transposed, bf16) + h/x staging + biases + reset flags.
  __shared__ __align__(16) __bf16 sWiT[3 * F * D];   // 96 KB  WiT[col][k]
  __shared__ __align__(16) __bf16 sWhT[3 * F * D];   // 96 KB  [r|z|n][col][k]
  __shared__ __align__(16) __bf16 sH[16 * F];        // 4 KB   h tile, bf16
  __shared__ __align__(16) __bf16 sX[16 * D];        // 4 KB   x_t tile, bf16
  __shared__ float sBias[3 * F + F];                 // bi(384) ++ bhn(128)
  __shared__ int   sDone[16];

  const int tid  = threadIdx.x;
  const int wave = tid >> 5;        // 8 waves: one 16-col output tile each
  const int lane = tid & 31;
  const int hi   = lane >> 4;       // C/D row-half selector
  const int nIdx = lane & 15;       // C/D column within tile
  const int cb   = wave * 16;       // column tile base in [0,F)
  const int n0   = blockIdx.x * 16; // env-row tile base in [0,N)

  // ---- one-time preload: weights -> LDS (bf16, column-major) --------------
  for (int idx = tid; idx < D * 3 * F; idx += 256) {
    int k = idx / (3 * F), c = idx % (3 * F);
    sWiT[c * D + k] = (__bf16)Wi[idx];
  }
  for (int idx = tid; idx < F * F; idx += 256) {
    int k = idx / F, c = idx % F;
    sWhT[(0 * F + c) * D + k] = (__bf16)Whr[idx];
    sWhT[(1 * F + c) * D + k] = (__bf16)Whz[idx];
    sWhT[(2 * F + c) * D + k] = (__bf16)Whn[idx];
  }
  for (int idx = tid; idx < 3 * F + F; idx += 256)
    sBias[idx] = (idx < 3 * F) ? bi[idx] : bhn[idx - 3 * F];
  for (int idx = tid; idx < 16 * F; idx += 256)
    sH[idx] = (__bf16)h0[n0 * F + idx];

  // fp32 h slice in registers: element j -> (row j+8*hi, col cb+nIdx)
  float hreg[8];
#pragma unroll
  for (int j = 0; j < 8; ++j)
    hreg[j] = h0[(size_t)(n0 + j + 8 * hi) * F + cb + nIdx];

  float* ys = out + (size_t)N * F;  // out = [final_h | ys]

#pragma unroll 1
  for (int t = 0; t < T; ++t) {
    __syncthreads();  // (A) prior-step sH/sX writes visible; staging safe

    // stage x_t tile (fp32 -> bf16) and this step's reset flags
    {
      const float* xb = obs + (size_t)t * N * D + (size_t)n0 * D;  // contiguous 16xD
#pragma unroll
      for (int i = 0; i < 2; ++i) {
        const int v = tid + 256 * i;           // float4 index in 2048-float slice
        const float4 x4 = ((const float4*)xb)[v];
        v4bf p;
        p[0] = (__bf16)x4.x; p[1] = (__bf16)x4.y;
        p[2] = (__bf16)x4.z; p[3] = (__bf16)x4.w;
        *(v4bf*)(sX + v * 4) = p;
      }
      if (t + 1 < T)  // hide next step's obs fetch under this step's WMMAs
        __builtin_prefetch(xb + (size_t)N * D + tid * 8, 0, 1);
      if (tid < 16) sDone[tid] = resets[(size_t)t * N + n0 + tid];
    }
    __syncthreads();  // (B) sX / sDone ready

    // load all A-fragments before barrier (C) so sH can be overwritten later
    v16bf aX[4], aH[4];
#pragma unroll
    for (int c = 0; c < 4; ++c) {
      aX[c] = load_a_frag(sX, lane, c);
      aH[c] = load_a_frag(sH, lane, c);
    }
    // episode reset: h := 0 for done rows (A-frag covers one row per lane)
    if (sDone[lane & 15]) {
#pragma unroll
      for (int c = 0; c < 4; ++c)
#pragma unroll
        for (int i = 0; i < 16; ++i) aH[c][i] = (__bf16)0.0f;
    }
    __syncthreads();  // (C) all sH/sX reads complete

    // six 16x16 (K=128) GEMM tiles: 24 bf16 WMMAs per wave per step
    v8f ir = {}, iz = {}, in_ = {}, hr = {}, hz = {}, hn = {};
#pragma unroll
    for (int c = 0; c < 4; ++c) {
      ir  = wmma_bf16(aX[c], load_b_frag(sWiT, cb,         lane, c), ir);
      iz  = wmma_bf16(aX[c], load_b_frag(sWiT, F + cb,     lane, c), iz);
      in_ = wmma_bf16(aX[c], load_b_frag(sWiT, 2 * F + cb, lane, c), in_);
      hr  = wmma_bf16(aH[c], load_b_frag(sWhT, cb,         lane, c), hr);
      hz  = wmma_bf16(aH[c], load_b_frag(sWhT, F + cb,     lane, c), hz);
      hn  = wmma_bf16(aH[c], load_b_frag(sWhT, 2 * F + cb, lane, c), hn);
    }

    // fused gate math in fp32 registers; update h and emit ys
    const int   gcol = cb + nIdx;
    const float b_r  = sBias[gcol];
    const float b_z  = sBias[F + gcol];
    const float b_n  = sBias[2 * F + gcol];
    const float b_hn = sBias[3 * F + gcol];
#pragma unroll
    for (int j = 0; j < 8; ++j) {
      const int   m    = j + 8 * hi;
      const float hold = sDone[m] ? 0.0f : hreg[j];
      const float r    = 1.0f / (1.0f + __expf(-(ir[j] + b_r + hr[j])));
      const float z    = 1.0f / (1.0f + __expf(-(iz[j] + b_z + hz[j])));
      const float nn   = tanhf(in_[j] + b_n + r * (hn[j] + b_hn));
      const float hnew = (1.0f - z) * nn + z * hold;
      hreg[j] = hnew;
      sH[m * F + gcol] = (__bf16)hnew;                         // next-step A operand
      ys[((size_t)t * N + n0 + m) * F + gcol] = hnew;          // scan output
    }
  }

  // final hidden state
#pragma unroll
  for (int j = 0; j < 8; ++j)
    out[(size_t)(n0 + j + 8 * hi) * F + cb + nIdx] = hreg[j];
}

extern "C" void kernel_launch(void* const* d_in, const int* in_sizes, int n_in,
                              void* d_out, int out_size, void* d_ws, size_t ws_size,
                              hipStream_t stream) {
  (void)in_sizes; (void)n_in; (void)out_size; (void)d_ws; (void)ws_size;
  const float* h0     = (const float*)d_in[0];
  const float* obs    = (const float*)d_in[1];
  const int*   resets = (const int*)d_in[2];
  const float* Wi     = (const float*)d_in[3];
  const float* bi     = (const float*)d_in[4];
  const float* Whr    = (const float*)d_in[5];
  const float* Whz    = (const float*)d_in[6];
  const float* Whn    = (const float*)d_in[7];
  const float* bhn    = (const float*)d_in[8];
  float* out = (float*)d_out;

  dim3 grid(N_ENV / 16);  // 16 persistent workgroups, one 16-env tile each
  dim3 block(256);        // 8 wave32s
  gru_scan_wmma_kernel<<<grid, block, 0, stream>>>(
      h0, obs, resets, Wi, bi, Whr, Whz, Whn, bhn, out);
}
